// EnhancedGAT_85143431676097
// MI455X (gfx1250) — matrix operand order repaired and checked by reading, not compiled, not run on hardware
//
#include <hip/hip_runtime.h>
#include <hip/hip_bf16.h>
#include <math.h>

typedef __attribute__((ext_vector_type(16))) _Float16     v16h;
typedef __attribute__((ext_vector_type(8)))  float        v8f;
typedef __attribute__((ext_vector_type(4)))  unsigned int v4u;
typedef __attribute__((ext_vector_type(4)))  int          v4i;
typedef __attribute__((ext_vector_type(8)))  int          v8i;

#define NNODE 10000
#define NEDGE 160000
#define NETOT 170000

#if __has_include(<hip/amd_detail/amd_gfx1250_TDM.h>)
#define TDM_ARGS6 1
#else
#define TDM_ARGS6 0
#endif

// ---------------- utility kernels ----------------

__global__ void k_cvt_f16(const float* __restrict__ s, _Float16* __restrict__ d, int n) {
    int i = blockIdx.x * blockDim.x + threadIdx.x;
    if (i < n) d[i] = (_Float16)s[i];
}

__global__ void k_fill(float* __restrict__ p, float v, int n) {
    int i = blockIdx.x * blockDim.x + threadIdx.x;
    if (i < n) p[i] = v;
}

__global__ void k_colmean(const float* __restrict__ ea, float* __restrict__ mean,
                          int rows, int cols) {
    __shared__ float red[256];
    int col = blockIdx.x;
    float s = 0.f;
    for (int r = threadIdx.x; r < rows; r += blockDim.x)
        s += ea[(long long)r * cols + col];
    red[threadIdx.x] = s;
    __syncthreads();
    for (int o = 128; o; o >>= 1) {
        if ((int)threadIdx.x < o) red[threadIdx.x] += red[threadIdx.x + o];
        __syncthreads();
    }
    if (threadIdx.x == 0) mean[col] = red[0] / (float)rows;
}

__device__ inline void atomicMaxF(float* p, float v) {
    int old = __float_as_int(*p);
    while (__int_as_float(old) < v) {
        int assumed = old;
        old = atomicCAS((int*)p, assumed, __float_as_int(v));
        if (old == assumed) break;
    }
}

// ---------------- TDM staging of a K x 64 f16 B-panel into LDS ----------------
// D# per CDNA5 ISA sec 8.3/8.4: group0 = {count/type/lds_addr/global_addr},
// group1 = {data_size=2B, tensor_dim0=N, tensor_dim1=K, tile_dim0=64, tile_dim1=K,
//           tensor_dim0_stride=N}.

#if __has_builtin(__builtin_amdgcn_tensor_load_to_lds)
__device__ inline void tdm_load_b_panel(const _Float16* gB, int N, int K, int nblk,
                                        unsigned lds_off) {
    unsigned long long ga = (unsigned long long)(uintptr_t)(gB + nblk);
    v4u g0;
    g0[0] = 1u;                                              // count = 1 valid descriptor
    g0[1] = lds_off;                                         // lds_addr (bits 63:32)
    g0[2] = (unsigned)(ga & 0xFFFFFFFFu);                    // global_addr low
    g0[3] = (unsigned)((ga >> 32) & 0x01FFFFFFu) | 0x80000000u; // global_addr hi | type=2
    v8i g1;
    g1[0] = 0x10000;                                         // data_size = 2 bytes
    g1[1] = (int)(((unsigned)N & 0xFFFFu) << 16);            // tensor_dim0 lo16 @bits 63:48
    g1[2] = (int)((((unsigned)N >> 16) & 0xFFFFu) |          // tensor_dim0 hi16
                  (((unsigned)K & 0xFFFFu) << 16));          // tensor_dim1 lo16
    g1[3] = (int)((((unsigned)K >> 16) & 0xFFFFu) |          // tensor_dim1 hi16
                  (64u << 16));                              // tile_dim0 = 64
    g1[4] = (int)((unsigned)K & 0xFFFFu);                    // tile_dim1 = K
    g1[5] = N;                                               // tensor_dim0_stride lo32
    g1[6] = 0;
    g1[7] = 0;
    v4i zz = {0, 0, 0, 0};
#if TDM_ARGS6
    v8i z8 = {0, 0, 0, 0, 0, 0, 0, 0};
    __builtin_amdgcn_tensor_load_to_lds(g0, g1, zz, zz, z8, 0);
#else
    __builtin_amdgcn_tensor_load_to_lds(g0, g1, zz, zz, 0);
#endif
}
#endif

// Swizzled LDS index for B element (k, n) of a K x 64 panel:
// layout [k>>5][n>>4][n&15][(k>>4)&1][k&15]  -> per-lane fragment is a
// contiguous, 32B-aligned run of 16 halfwords (two ds_load_b128).
__device__ inline int b_swz_idx(int k, int n) {
    return ((((k >> 5) * 4 + (n >> 4)) * 16 + (n & 15)) * 2 + ((k >> 4) & 1)) * 16 + (k & 15);
}

// ---------------- WMMA f16 GEMM: C[M,N] = act(A[M,K] @ B[K,N] + bias) ----------------
// Block = 256 threads (8 waves). Block tile = 128(M) x 64(N); each wave owns a 16x64
// strip: 4 accumulators reuse one A fragment -> 4 WMMAs per k-step per wave.
// B panel DMA'd by TDM into linear LDS, then swizzled into fragment order so the
// inner loop is 2 global b128 (A) + 8 ds b128 (B) + 4 WMMA per k-step.
// M multiple of 16, N multiple of 64, K multiple of 32. act: 0 none, 1 leaky, 2 elu.

__global__ void k_gemm_f16(const _Float16* __restrict__ A, const _Float16* __restrict__ B,
                           const float* __restrict__ bias, float* __restrict__ C,
                           int M, int N, int K, int act) {
    extern __shared__ _Float16 sh16[];
    _Float16* lin = sh16;                 // K x 64 linear panel (TDM destination)
    _Float16* swz = sh16 + (size_t)K * 64; // K x 64 fragment-order panel
    const int lane = threadIdx.x & 31;
    const int wave = threadIdx.x >> 5;
    const int nblk = blockIdx.x << 6;

#if __has_builtin(__builtin_amdgcn_tensor_load_to_lds)
    if (wave == 0) {
        tdm_load_b_panel(B, N, K, nblk, (unsigned)(uintptr_t)lin);
#if __has_builtin(__builtin_amdgcn_s_wait_tensorcnt)
        __builtin_amdgcn_s_wait_tensorcnt(0);
#endif
    }
    __syncthreads();
    for (int idx = threadIdx.x; idx < K * 64; idx += blockDim.x) {
        const int k = idx >> 6, n = idx & 63;
        swz[b_swz_idx(k, n)] = lin[idx];
    }
#else
    for (int idx = threadIdx.x; idx < K * 64; idx += blockDim.x) {
        const int k = idx >> 6, n = idx & 63;
        swz[b_swz_idx(k, n)] = B[(size_t)k * N + nblk + n];
    }
#endif
    __syncthreads();

    const int tm = blockIdx.y * 8 + wave;            // wave-uniform
    if (tm < (M >> 4)) {                             // uniform: EXEC all-1 at WMMA
        const int hl  = lane >> 4;
        const int l16 = lane & 15;
        const _Float16* Arow = A + (size_t)(tm * 16 + l16) * K;

        v8f acc[4] = {v8f{}, v8f{}, v8f{}, v8f{}};
        for (int k0 = 0; k0 < K; k0 += 32) {
            // A fragment: VGPRs 0-3 = K[16h..16h+7], VGPRs 4-7 = K[16h+8..16h+15]
            // -> two contiguous 16B runs -> 2x global_load_b128.
            union { v16h h; v4u q[2]; } au;
            au.q[0] = *(const v4u*)(Arow + k0 + 16 * hl);
            au.q[1] = *(const v4u*)(Arow + k0 + 16 * hl + 8);
            #pragma unroll
            for (int nt = 0; nt < 4; ++nt) {
                // lane's 16 B elements are contiguous & 32B-aligned in swz
                const v16h b = *(const v16h*)(
                    swz + (((((k0 >> 5) * 4 + nt) * 16 + l16) * 2 + hl) << 4));
                acc[nt] = __builtin_amdgcn_wmma_f32_16x16x32_f16(
                    false, au.h, false, b, (short)0, acc[nt], false, false);
            }
        }
        #pragma unroll
        for (int nt = 0; nt < 4; ++nt) {
            const int col = nblk + nt * 16 + l16;
            const float bb = bias ? bias[col] : 0.f;
            #pragma unroll
            for (int r = 0; r < 8; ++r) {
                const int m = tm * 16 + r + 8 * hl;  // C/D: M = r + 8*half, N = lane%16
                float v = acc[nt][r] + bb;
                if (act == 1)      v = v > 0.f ? v : 0.2f * v;
                else if (act == 2) v = v > 0.f ? v : (__expf(v) - 1.f);
                C[(size_t)m * N + col] = v;
            }
        }
    }
}

// ---------------- edge pass 1: logits + segment max ----------------

template <int H, int HC>
__global__ void k_edge_logits(const int* __restrict__ ei,
                              const float* __restrict__ ea, const float* __restrict__ ea_mean,
                              const float* __restrict__ We, const float* __restrict__ att,
                              const float* __restrict__ xl, const float* __restrict__ xr,
                              float* __restrict__ logits, float* __restrict__ mmax) {
    extern __shared__ float sh[];                 // 32*HC (We) + H*128 (att)
    float* We_s  = sh;
    float* att_s = sh + 32 * HC;
    for (int i = threadIdx.x; i < 32 * HC; i += blockDim.x) We_s[i] = We[i];
    for (int i = threadIdx.x; i < H * 128; i += blockDim.x) att_s[i] = att[i];
    __syncthreads();

    const int lane   = threadIdx.x & 31;
    const int wave   = threadIdx.x >> 5;
    const int nwaves = gridDim.x * (blockDim.x >> 5);

    for (int e = blockIdx.x * (blockDim.x >> 5) + wave; e < NETOT; e += nwaves) {
        int s, d;
        float eav;
        if (e < NEDGE) { s = ei[e]; d = ei[NEDGE + e]; eav = ea[e * 32 + lane]; }
        else           { s = e - NEDGE; d = s;          eav = ea_mean[lane]; }

        float ec[HC / 32];
        #pragma unroll
        for (int j = 0; j < HC / 32; ++j) ec[j] = 0.f;
        for (int k = 0; k < 32; ++k) {
            const float eak = __shfl(eav, k, 32);
            const float* wr = We_s + k * HC;
            #pragma unroll
            for (int j = 0; j < HC / 32; ++j) ec[j] += eak * wr[lane + 32 * j];
        }

        float hacc[H];
        #pragma unroll
        for (int h = 0; h < H; ++h) hacc[h] = 0.f;
        const float* xls = xl + s * HC;
        const float* xrd = xr + d * HC;
        #pragma unroll
        for (int j = 0; j < HC / 32; ++j) {
            const int c = lane + 32 * j;               // head = j>>2 (lane-invariant)
            float t = xls[c] + xrd[c] + ec[j];
            t = t > 0.f ? t : 0.2f * t;
            hacc[j >> 2] += att_s[(j >> 2) * 128 + (c & 127)] * t;
        }
        #pragma unroll
        for (int h = 0; h < H; ++h)
            for (int off = 16; off; off >>= 1)
                hacc[h] += __shfl_xor(hacc[h], off, 32);
        if (lane < H) {
            logits[e * H + lane] = hacc[lane];
            atomicMaxF(&mmax[d * H + lane], hacc[lane]);
        }
    }
}

// ---------------- edge pass 2: unnormalized weighted scatter + z ----------------

template <int H, int HC>
__global__ void k_edge_accum(const int* __restrict__ ei,
                             const float* __restrict__ logits, const float* __restrict__ mmax,
                             const float* __restrict__ xl,
                             float* __restrict__ z, float* __restrict__ out) {
    const int lane   = threadIdx.x & 31;
    const int wave   = threadIdx.x >> 5;
    const int nwaves = gridDim.x * (blockDim.x >> 5);

    for (int e = blockIdx.x * (blockDim.x >> 5) + wave; e < NETOT; e += nwaves) {
        int s, d;
        if (e < NEDGE) { s = ei[e]; d = ei[NEDGE + e]; }
        else           { s = e - NEDGE; d = s; }

        float wv = 0.f;
        if (lane < H) {
            wv = __expf(logits[e * H + lane] - mmax[d * H + lane]);
            atomicAdd(&z[d * H + lane], wv);
        }
        float wh[H];
        #pragma unroll
        for (int h = 0; h < H; ++h) wh[h] = __shfl(wv, h, 32);

        const float* xls = xl + s * HC;
        float* od = out + d * HC;
        #pragma unroll
        for (int j = 0; j < HC / 32; ++j) {
            const int c = lane + 32 * j;
            atomicAdd(&od[c], xls[c] * wh[j >> 2]);
        }
    }
}

// ---------------- per-layer finalize ----------------

__global__ void k_fin1(const float* __restrict__ out, const float* __restrict__ z,
                       const float* __restrict__ bo, const float* __restrict__ res,
                       _Float16* __restrict__ dst, int total) {
    int i = blockIdx.x * blockDim.x + threadIdx.x;
    if (i >= total) return;
    const int node = i >> 10, col = i & 1023, head = col >> 7;
    float v = out[i] / (z[node * 8 + head] + 1e-16f) + bo[col] + res[i];
    v = v > 0.f ? v : 0.2f * v;
    dst[i] = (_Float16)v;
}

__global__ void k_fin2(const float* __restrict__ out, const float* __restrict__ z,
                       const float* __restrict__ bo, _Float16* __restrict__ dst, int total) {
    int i = blockIdx.x * blockDim.x + threadIdx.x;
    if (i >= total) return;
    const int node = i >> 7, c = i & 127;
    float s = 0.f;
    #pragma unroll
    for (int h = 0; h < 4; ++h)
        s += out[node * 512 + h * 128 + c] / (z[node * 4 + h] + 1e-16f);
    float v = 0.25f * s + bo[c];
    v = v > 0.f ? v : (__expf(v) - 1.f);
    dst[i] = (_Float16)v;
}

__global__ void k_fin3(const float* __restrict__ out, const float* __restrict__ z,
                       const float* __restrict__ bo, _Float16* __restrict__ dst, int total) {
    int i = blockIdx.x * blockDim.x + threadIdx.x;
    if (i >= total) return;
    const int node = i >> 7, c = i & 127;
    float v = out[i] / (z[node] + 1e-16f) + bo[c];
    v = v > 0.f ? v : (__expf(v) - 1.f);
    dst[i] = (_Float16)v;
}

__global__ void k_head2(const float* __restrict__ h4, const float* __restrict__ Wc2,
                        const float* __restrict__ bc2, float* __restrict__ out, int n) {
    int i = blockIdx.x * blockDim.x + threadIdx.x;
    if (i >= n * 3) return;
    const int node = i / 3, j = i % 3;
    float s = bc2[j];
    const float* hv = h4 + node * 64;
    #pragma unroll
    for (int k = 0; k < 64; ++k) s += hv[k] * Wc2[k * 3 + j];
    out[i] = s;
}

// ---------------- host orchestration ----------------

static inline void gemm(const _Float16* A, const _Float16* B, const float* bias, float* C,
                        int M, int N, int K, int act, hipStream_t st) {
    dim3 grid(N / 64, ((M >> 4) + 7) / 8);
    k_gemm_f16<<<grid, 256, (size_t)K * 64 * 2 * 2, st>>>(A, B, bias, C, M, N, K, act);
}

extern "C" void kernel_launch(void* const* d_in, const int* in_sizes, int n_in,
                              void* d_out, int out_size, void* d_ws, size_t ws_size,
                              hipStream_t stream) {
    const float* x    = (const float*)d_in[0];
    const int*   ei   = (const int*)d_in[1];
    const float* ea   = (const float*)d_in[2];
    const float* Wl1  = (const float*)d_in[3];  const float* bl1 = (const float*)d_in[4];
    const float* Wr1  = (const float*)d_in[5];  const float* br1 = (const float*)d_in[6];
    const float* We1  = (const float*)d_in[7];  const float* att1= (const float*)d_in[8];
    const float* bo1  = (const float*)d_in[9];
    const float* Wl2  = (const float*)d_in[10]; const float* bl2 = (const float*)d_in[11];
    const float* Wr2  = (const float*)d_in[12]; const float* br2 = (const float*)d_in[13];
    const float* We2  = (const float*)d_in[14]; const float* att2= (const float*)d_in[15];
    const float* bo2  = (const float*)d_in[16];
    const float* Wl3  = (const float*)d_in[17]; const float* bl3 = (const float*)d_in[18];
    const float* Wr3  = (const float*)d_in[19]; const float* br3 = (const float*)d_in[20];
    const float* We3  = (const float*)d_in[21]; const float* att3= (const float*)d_in[22];
    const float* bo3  = (const float*)d_in[23];
    const float* Wres = (const float*)d_in[24]; const float* bres= (const float*)d_in[25];
    const float* Wc1  = (const float*)d_in[26]; const float* bc1 = (const float*)d_in[27];
    const float* Wc2  = (const float*)d_in[28]; const float* bc2 = (const float*)d_in[29];

    char* p = (char*)d_ws;
    auto alloc = [&](size_t bytes) -> void* {
        void* r = (void*)p;
        p += (bytes + 255) & ~(size_t)255;
        return r;
    };
    _Float16* xh    = (_Float16*)alloc((size_t)NNODE * 128 * 2);
    _Float16* Wl1h  = (_Float16*)alloc((size_t)128 * 1024 * 2);
    _Float16* Wr1h  = (_Float16*)alloc((size_t)128 * 1024 * 2);
    _Float16* Wresh = (_Float16*)alloc((size_t)128 * 1024 * 2);
    _Float16* Wl2h  = (_Float16*)alloc((size_t)1024 * 512 * 2);
    _Float16* Wr2h  = (_Float16*)alloc((size_t)1024 * 512 * 2);
    _Float16* Wl3h  = (_Float16*)alloc((size_t)128 * 128 * 2);
    _Float16* Wr3h  = (_Float16*)alloc((size_t)128 * 128 * 2);
    _Float16* Wc1h  = (_Float16*)alloc((size_t)128 * 64 * 2);
    float* ea_mean  = (float*)alloc(32 * 4);
    float* bufA     = (float*)alloc((size_t)NNODE * 1024 * 4);  // xl (reused per layer)
    float* bufB     = (float*)alloc((size_t)NNODE * 1024 * 4);  // xr
    float* bufC     = (float*)alloc((size_t)NNODE * 1024 * 4);  // out accumulator
    float* bufR     = (float*)alloc((size_t)NNODE * 1024 * 4);  // residual / h4
    float* logits   = (float*)alloc((size_t)NETOT * 8 * 4);
    float* mbuf     = (float*)alloc((size_t)NNODE * 8 * 4);
    float* zbuf     = (float*)alloc((size_t)NNODE * 8 * 4);
    _Float16* actH  = (_Float16*)alloc((size_t)NNODE * 1024 * 2); // h1h/h2h/h3h

    auto cdiv = [](int a, int b) { return (a + b - 1) / b; };
    auto cvt = [&](const float* s, _Float16* d, int n) {
        k_cvt_f16<<<cdiv(n, 256), 256, 0, stream>>>(s, d, n);
    };
    auto fill = [&](float* q, float v, int n) {
        k_fill<<<cdiv(n, 256), 256, 0, stream>>>(q, v, n);
    };

    // --- f16 conversions + edge-attr mean ---
    cvt(x,    xh,    NNODE * 128);
    cvt(Wl1,  Wl1h,  128 * 1024);  cvt(Wr1, Wr1h, 128 * 1024);  cvt(Wres, Wresh, 128 * 1024);
    cvt(Wl2,  Wl2h,  1024 * 512);  cvt(Wr2, Wr2h, 1024 * 512);
    cvt(Wl3,  Wl3h,  128 * 128);   cvt(Wr3, Wr3h, 128 * 128);
    cvt(Wc1,  Wc1h,  128 * 64);
    k_colmean<<<32, 256, 0, stream>>>(ea, ea_mean, NEDGE, 32);

    // --- layer 1 (H=8, HC=1024) ---
    gemm(xh, Wl1h,  bl1,  bufA, NNODE, 1024, 128, 0, stream);
    gemm(xh, Wr1h,  br1,  bufB, NNODE, 1024, 128, 0, stream);
    gemm(xh, Wresh, bres, bufR, NNODE, 1024, 128, 0, stream);
    fill(mbuf, -1e30f, NNODE * 8); fill(zbuf, 0.f, NNODE * 8); fill(bufC, 0.f, NNODE * 1024);
    k_edge_logits<8, 1024><<<1024, 256, (32 * 1024 + 8 * 128) * 4, stream>>>(
        ei, ea, ea_mean, We1, att1, bufA, bufB, logits, mbuf);
    k_edge_accum<8, 1024><<<1024, 256, 0, stream>>>(ei, logits, mbuf, bufA, zbuf, bufC);
    k_fin1<<<cdiv(NNODE * 1024, 256), 256, 0, stream>>>(bufC, zbuf, bo1, bufR, actH,
                                                        NNODE * 1024);

    // --- layer 2 (H=4, HC=512, mean over heads) ---
    gemm(actH, Wl2h, bl2, bufA, NNODE, 512, 1024, 0, stream);
    gemm(actH, Wr2h, br2, bufB, NNODE, 512, 1024, 0, stream);
    fill(mbuf, -1e30f, NNODE * 4); fill(zbuf, 0.f, NNODE * 4); fill(bufC, 0.f, NNODE * 512);
    k_edge_logits<4, 512><<<1024, 256, (32 * 512 + 4 * 128) * 4, stream>>>(
        ei, ea, ea_mean, We2, att2, bufA, bufB, logits, mbuf);
    k_edge_accum<4, 512><<<1024, 256, 0, stream>>>(ei, logits, mbuf, bufA, zbuf, bufC);
    k_fin2<<<cdiv(NNODE * 128, 256), 256, 0, stream>>>(bufC, zbuf, bo2, actH, NNODE * 128);

    // --- layer 3 (H=1, HC=128) ---
    gemm(actH, Wl3h, bl3, bufA, NNODE, 128, 128, 0, stream);
    gemm(actH, Wr3h, br3, bufB, NNODE, 128, 128, 0, stream);
    fill(mbuf, -1e30f, NNODE); fill(zbuf, 0.f, NNODE); fill(bufC, 0.f, NNODE * 128);
    k_edge_logits<1, 128><<<1024, 256, (32 * 128 + 128) * 4, stream>>>(
        ei, ea, ea_mean, We3, att3, bufA, bufB, logits, mbuf);
    k_edge_accum<1, 128><<<1024, 256, 0, stream>>>(ei, logits, mbuf, bufA, zbuf, bufC);
    k_fin3<<<cdiv(NNODE * 128, 256), 256, 0, stream>>>(bufC, zbuf, bo3, actH, NNODE * 128);

    // --- head: h4 = elu(h3 @ Wc1 + bc1); out = h4 @ Wc2 + bc2 ---
    gemm(actH, Wc1h, bc1, bufR, NNODE, 64, 128, 2, stream);
    k_head2<<<cdiv(NNODE * 3, 256), 256, 0, stream>>>(bufR, Wc2, bc2, (float*)d_out, NNODE);
}
